// CausalSelfAttention_7662221656822
// MI455X (gfx1250) — matrix operand order repaired and checked by reading, not compiled, not run on hardware
//
#include <hip/hip_runtime.h>
#include <stdint.h>
#include <math.h>

// ---------------------------------------------------------------------------
// Causal self-attention for MI455X (gfx1250, wave32, WMMA, async global->LDS).
//   1) convert x / W_qkv / W_proj fp32 -> bf16 (RNE)
//   2) GEMM (bf16 WMMA, f32 acc, double-buffered async LDS tiles):
//        qkv = x @ W_qkv + b ; scatter Q/K/V [B,H,T,64] bf16
//   3) flash attention per (b,h): S = Q K^T (WMMA), causal online softmax,
//        O = P V (WMMA); K/V tiles double-buffered via async DMA
//   4) GEMM: out = att @ W_proj + b  (fp32 output)
// ---------------------------------------------------------------------------

#define BATCH   4
#define TSEQ    2048
#define DMODEL  1024
#define NHEAD   16
#define HDIM    64
#define MROWS   (BATCH * TSEQ)   // 8192

typedef __attribute__((ext_vector_type(16))) __bf16    bf16x16;
typedef __attribute__((ext_vector_type(8)))  float     f32x8;
typedef __attribute__((ext_vector_type(8)))  uint32_t  u32x8;

__device__ __forceinline__ uint16_t f32_to_bf16(float f) {
  uint32_t u = __builtin_bit_cast(uint32_t, f);
  uint32_t r = u + 0x7FFFu + ((u >> 16) & 1u);   // round-to-nearest-even
  return (uint16_t)(r >> 16);
}

__device__ __forceinline__ f32x8 wmma_bf16(const u32x8& a, const u32x8& b, const f32x8& c) {
  // v_wmma_f32_16x16x32_bf16 : D = A(16x32) * B(32x16) + C(16x16 f32)
  return __builtin_amdgcn_wmma_f32_16x16x32_bf16(
      false, __builtin_bit_cast(bf16x16, a),
      false, __builtin_bit_cast(bf16x16, b),
      (short)0, c, false, false);
}

// Async DMA: global (16B) -> LDS, tracked by ASYNCcnt, no VGPR staging.
// LDS byte address = low 32 bits of the generic pointer (shared aperture is
// 2^32-aligned: flat LDS mapping uses addr[31:0]).
__device__ __forceinline__ void async_copy_b128(const void* lds_dst, const void* gsrc) {
  uint32_t l = (uint32_t)(uintptr_t)lds_dst;
  uint64_t g = (uint64_t)(uintptr_t)gsrc;
  asm volatile("global_load_async_to_lds_b128 %0, %1, off"
               :: "v"(l), "v"(g) : "memory");
}
__device__ __forceinline__ void wait_async0() {
  asm volatile("s_wait_asynccnt 0x0" ::: "memory");
}
__device__ __forceinline__ void wait_ds0() {
  asm volatile("s_wait_dscnt 0x0" ::: "memory");
}

// A-matrix 16x32 bf16 fragment from LDS tile stored row-major [m][k].
// Layout: lanes 0-15 (m=lane): V0..7 -> K=0..15
//         lanes 16-31:          V0..7 -> K=16..31
__device__ __forceinline__ u32x8 load_a_frag(const uint16_t* t, int pitch, int row0, int k0, int lane) {
  int m = row0 + (lane & 15);
  int g = (lane >> 4) & 1;
  const uint16_t* p = t + m * pitch + k0 + 16 * g;
  u32x8 r;
#pragma unroll
  for (int v = 0; v < 4; ++v) r[v]     = *(const uint32_t*)(p + 2 * v);
#pragma unroll
  for (int v = 0; v < 4; ++v) r[4 + v] = *(const uint32_t*)(p + 8 + 2 * v);
  return r;
}

// B-matrix 32x16 fragment where the LDS tile holds B^T: [n][k], k contiguous.
__device__ __forceinline__ u32x8 load_bT_frag(const uint16_t* t, int pitch, int n0, int k0, int lane) {
  int n = n0 + (lane & 15);
  int g = (lane >> 4) & 1;
  const uint16_t* p = t + n * pitch + k0 + 16 * g;
  u32x8 r;
#pragma unroll
  for (int v = 0; v < 8; ++v) r[v] = *(const uint32_t*)(p + 2 * v);
  return r;
}

// B-matrix 32x16 fragment where the LDS tile is [k][n], n contiguous.
__device__ __forceinline__ u32x8 load_b_frag_kn(const uint16_t* t, int pitch, int n0, int k0, int lane) {
  int n = n0 + (lane & 15);
  int g = (lane >> 4) & 1;
  const uint16_t* p = t + (k0 + 16 * g) * pitch + n;
  u32x8 r;
#pragma unroll
  for (int v = 0; v < 8; ++v) {
    uint32_t lo = p[(2 * v) * pitch];
    uint32_t hi = p[(2 * v + 1) * pitch];
    r[v] = lo | (hi << 16);
  }
  return r;
}

// ---------------------------------------------------------------------------
// fp32 -> bf16 convert
// ---------------------------------------------------------------------------
__global__ void cvt_bf16_kernel(const float* __restrict__ in, uint16_t* __restrict__ out, int n) {
  int i = blockIdx.x * blockDim.x + threadIdx.x;
  if (i < n) out[i] = f32_to_bf16(in[i]);
}

// ---------------------------------------------------------------------------
// Tiled bf16 WMMA GEMM: C[M,N] = A[M,K] * B[K,N] + bias[N]
// 256 threads = 8 waves; block tile 128x128, K-slab 32, double-buffered
// async global->LDS tiles. Wave (wm,wn) computes a 64x32 sub-tile (4x2 WMMA).
// MODE 0: scatter Q/K/V bf16 [B,H,T,64] ; MODE 1: fp32 row-major output.
// ---------------------------------------------------------------------------
template <int MODE>
__global__ __launch_bounds__(256) void gemm_bf16_kernel(
    const uint16_t* __restrict__ A, const uint16_t* __restrict__ Bw,
    const float* __restrict__ bias, float* __restrict__ outF,
    uint16_t* __restrict__ outQ, uint16_t* __restrict__ outK, uint16_t* __restrict__ outV,
    int M, int N, int K) {
  constexpr int AP = 40;    // A tile [128][32] pitch (ushorts)
  constexpr int BP = 136;   // B tile [32][128] pitch (ushorts), natural [k][n]
  __shared__ uint16_t At[2][128 * AP];
  __shared__ uint16_t Bt[2][32 * BP];

  const int tid  = threadIdx.x;
  const int lane = tid & 31;
  const int w    = tid >> 5;
  const int wm   = w >> 2;       // 0..1
  const int wn   = w & 3;        // 0..3
  const int m0   = blockIdx.y * 128;
  const int n0   = blockIdx.x * 128;

  auto issue_tiles = [&](int buf, int k0) {
    // A: 128 rows x 32 K = 512 x 16B chunks, 2 per thread
#pragma unroll
    for (int i = 0; i < 2; ++i) {
      int u = tid + i * 256;
      int r = u >> 2, c = (u & 3) * 8;
      async_copy_b128(&At[buf][r * AP + c], &A[(size_t)(m0 + r) * K + k0 + c]);
    }
    // B: 32 rows x 128 N = 512 x 16B chunks, 2 per thread (no transpose needed)
#pragma unroll
    for (int i = 0; i < 2; ++i) {
      int u = tid + i * 256;
      int r = u >> 4, c = (u & 15) * 8;
      async_copy_b128(&Bt[buf][r * BP + c], &Bw[(size_t)(k0 + r) * N + n0 + c]);
    }
  };

  f32x8 acc[4][2];
#pragma unroll
  for (int i = 0; i < 4; ++i)
#pragma unroll
    for (int j = 0; j < 2; ++j) acc[i][j] = {};

  issue_tiles(0, 0);
  wait_async0();
  __syncthreads();

  const int nslab = K / 32;
  for (int s = 0; s < nslab; ++s) {
    const int cur = s & 1;
    if (s + 1 < nslab) issue_tiles(cur ^ 1, (s + 1) * 32);   // DMA overlaps WMMA

    u32x8 af[4];
#pragma unroll
    for (int i = 0; i < 4; ++i)
      af[i] = load_a_frag(&At[cur][0], AP, wm * 64 + i * 16, 0, lane);
#pragma unroll
    for (int j = 0; j < 2; ++j) {
      u32x8 bf = load_b_frag_kn(&Bt[cur][0], BP, wn * 32 + j * 16, 0, lane);
#pragma unroll
      for (int i = 0; i < 4; ++i) acc[i][j] = wmma_bf16(af[i], bf, acc[i][j]);
    }

    wait_async0();
    __syncthreads();
  }

  // --- epilogue (C/D layout: row = r + 8*(lane/16), col = lane%16)
  const int g  = lane >> 4;
  const int nn = lane & 15;
#pragma unroll
  for (int i = 0; i < 4; ++i)
#pragma unroll
    for (int j = 0; j < 2; ++j)
#pragma unroll
      for (int r = 0; r < 8; ++r) {
        int gm = m0 + wm * 64 + i * 16 + r + 8 * g;
        int gn = n0 + wn * 32 + j * 16 + nn;
        float v = acc[i][j][r] + bias[gn];
        if (MODE == 0) {
          int s2 = gn >> 10;           // 0:q 1:k 2:v
          int c  = gn & 1023;
          int h = c >> 6, d = c & 63;
          int b = gm >> 11, t = gm & 2047;
          uint16_t* dst = (s2 == 0) ? outQ : (s2 == 1) ? outK : outV;
          dst[((((size_t)b * NHEAD + h) * TSEQ + t) << 6) + d] = f32_to_bf16(v);
        } else {
          outF[(size_t)gm * N + gn] = v;
        }
      }
}

// ---------------------------------------------------------------------------
// Flash attention (causal). grid = (T/64, B*H), 128 threads = 4 waves.
// Wave w owns 16 queries; K/V 64-key tiles double-buffered via async DMA.
// ---------------------------------------------------------------------------
__global__ __launch_bounds__(128) void attn_kernel(
    const uint16_t* __restrict__ Qb, const uint16_t* __restrict__ Kb,
    const uint16_t* __restrict__ Vb, uint16_t* __restrict__ attb) {
  constexpr int P = 72;                   // pitch for 64-wide bf16 tiles
  __shared__ uint16_t Qt[64 * P];
  __shared__ uint16_t Kt[2][64 * P];
  __shared__ uint16_t Vt[2][64 * P];
  __shared__ uint16_t Pt[4][16 * P];      // per-wave C->A relayout scratch

  const int tid   = threadIdx.x;
  const int lane  = tid & 31;
  const int w     = tid >> 5;             // wave 0..3
  const int qtile = blockIdx.x;           // 0..31
  const int bh    = blockIdx.y;           // 0..63
  const size_t base = (size_t)bh * TSEQ * HDIM;
  const int g  = lane >> 4;
  const int nn = lane & 15;
  const float scale = 0.125f;             // 1/sqrt(64)

  auto issue_kv = [&](int buf, int kt) {
    // 64 rows x 64 d = 512 x 16B chunks per tile, 4 per thread per tile
#pragma unroll
    for (int i = 0; i < 4; ++i) {
      int u = tid + i * 128;
      int r = u >> 3, c = (u & 7) * 8;
      size_t src = base + (size_t)(kt * 64 + r) * HDIM + c;
      async_copy_b128(&Kt[buf][r * P + c], &Kb[src]);
      async_copy_b128(&Vt[buf][r * P + c], &Vb[src]);
    }
  };

  // Q tile (64x64) + first K/V tile, all via async DMA
#pragma unroll
  for (int i = 0; i < 4; ++i) {
    int u = tid + i * 128;
    int r = u >> 3, c = (u & 7) * 8;
    async_copy_b128(&Qt[r * P + c], &Qb[base + (size_t)(qtile * 64 + r) * HDIM + c]);
  }
  issue_kv(0, 0);
  wait_async0();
  __syncthreads();

  float mrow[8], lrow[8];
  f32x8 of[4];
#pragma unroll
  for (int r = 0; r < 8; ++r) { mrow[r] = -INFINITY; lrow[r] = 0.f; }
#pragma unroll
  for (int j = 0; j < 4; ++j) of[j] = {};

  for (int kt = 0; kt <= qtile; ++kt) {
    const int cur = kt & 1;
    if (kt < qtile) issue_kv(cur ^ 1, kt + 1);     // DMA overlaps compute

    // S = Q(16x64) @ K^T(64x64) for this wave's queries
    f32x8 sacc[4];
#pragma unroll
    for (int j = 0; j < 4; ++j) sacc[j] = {};
#pragma unroll
    for (int ks = 0; ks < 2; ++ks) {
      u32x8 aq = load_a_frag(Qt, P, w * 16, ks * 32, lane);
#pragma unroll
      for (int j = 0; j < 4; ++j) {
        u32x8 bk = load_bT_frag(&Kt[cur][0], P, j * 16, ks * 32, lane);
        sacc[j] = wmma_bf16(aq, bk, sacc[j]);
      }
    }

    // scale + causal mask + row max
    float rmax[8];
#pragma unroll
    for (int r = 0; r < 8; ++r) rmax[r] = -INFINITY;
#pragma unroll
    for (int j = 0; j < 4; ++j)
#pragma unroll
      for (int r = 0; r < 8; ++r) {
        int q   = qtile * 64 + w * 16 + r + 8 * g;
        int key = kt * 64 + j * 16 + nn;
        float s = sacc[j][r] * scale;
        if (key > q) s = -INFINITY;
        sacc[j][r] = s;
        rmax[r] = fmaxf(rmax[r], s);
      }
#pragma unroll
    for (int off = 1; off < 16; off <<= 1)
#pragma unroll
      for (int r = 0; r < 8; ++r)
        rmax[r] = fmaxf(rmax[r], __shfl_xor(rmax[r], off, 32));

    float alpha[8], rsum[8];
#pragma unroll
    for (int r = 0; r < 8; ++r) {
      float mn = fmaxf(mrow[r], rmax[r]);
      alpha[r] = __expf(mrow[r] - mn);
      mrow[r]  = mn;
      rsum[r]  = 0.f;
    }

    // P = exp(S - m): stash bf16 in per-wave LDS (lane transpose C->A layout)
    uint16_t* pw = &Pt[w][0];
#pragma unroll
    for (int j = 0; j < 4; ++j)
#pragma unroll
      for (int r = 0; r < 8; ++r) {
        float p = __expf(sacc[j][r] - mrow[r]);
        rsum[r] += p;
        pw[(r + 8 * g) * P + j * 16 + nn] = f32_to_bf16(p);
      }
#pragma unroll
    for (int off = 1; off < 16; off <<= 1)
#pragma unroll
      for (int r = 0; r < 8; ++r)
        rsum[r] += __shfl_xor(rsum[r], off, 32);
#pragma unroll
    for (int r = 0; r < 8; ++r) lrow[r] = lrow[r] * alpha[r] + rsum[r];
#pragma unroll
    for (int j = 0; j < 4; ++j)
#pragma unroll
      for (int r = 0; r < 8; ++r) of[j][r] *= alpha[r];

    // DS ops are in-order per wave: drain, then read P back as A fragments
    wait_ds0();

    // O += P(16x64) @ V(64x64)
#pragma unroll
    for (int ks = 0; ks < 2; ++ks) {
      u32x8 ap = load_a_frag(pw, P, 0, ks * 32, lane);
#pragma unroll
      for (int j = 0; j < 4; ++j) {
        u32x8 bv = load_b_frag_kn(&Vt[cur][0], P, j * 16, ks * 32, lane);
        of[j] = wmma_bf16(ap, bv, of[j]);
      }
    }

    wait_async0();
    __syncthreads();
  }

  // normalize and emit bf16 att output [B,T,C] for the projection GEMM
  const int b = bh >> 4, h = bh & 15;
#pragma unroll
  for (int r = 0; r < 8; ++r) {
    float inv = 1.f / lrow[r];
    int t = qtile * 64 + w * 16 + r + 8 * g;
#pragma unroll
    for (int j = 0; j < 4; ++j) {
      int d = j * 16 + nn;
      attb[(size_t)(b * TSEQ + t) * DMODEL + h * HDIM + d] =
          f32_to_bf16(of[j][r] * inv);
    }
  }
}

// ---------------------------------------------------------------------------
// launch
// ---------------------------------------------------------------------------
extern "C" void kernel_launch(void* const* d_in, const int* in_sizes, int n_in,
                              void* d_out, int out_size, void* d_ws, size_t ws_size,
                              hipStream_t stream) {
  (void)in_sizes; (void)n_in; (void)out_size; (void)ws_size;
  const float* x     = (const float*)d_in[0];
  const float* Wqkv  = (const float*)d_in[1];
  const float* bqkv  = (const float*)d_in[2];
  const float* Wproj = (const float*)d_in[3];
  const float* bproj = (const float*)d_in[4];
  float* out = (float*)d_out;

  char* ws = (char*)d_ws;
  size_t off = 0;
  auto take = [&](size_t bytes) -> char* {
    char* p = ws + off;
    off += (bytes + 255) & ~(size_t)255;
    return p;
  };
  uint16_t* xb     = (uint16_t*)take((size_t)MROWS * DMODEL * 2);
  uint16_t* wqkvb  = (uint16_t*)take((size_t)DMODEL * 3 * DMODEL * 2);
  uint16_t* wprojb = (uint16_t*)take((size_t)DMODEL * DMODEL * 2);
  uint16_t* Qb     = (uint16_t*)take((size_t)MROWS * DMODEL * 2);
  uint16_t* Kbuf   = (uint16_t*)take((size_t)MROWS * DMODEL * 2);
  uint16_t* Vbuf   = (uint16_t*)take((size_t)MROWS * DMODEL * 2);
  uint16_t* attb   = (uint16_t*)take((size_t)MROWS * DMODEL * 2);

  const int nx  = MROWS * DMODEL;         // 8388608
  const int nw1 = DMODEL * 3 * DMODEL;    // 3145728
  const int nw2 = DMODEL * DMODEL;        // 1048576
  cvt_bf16_kernel<<<(nx  + 255) / 256, 256, 0, stream>>>(x,     xb,     nx);
  cvt_bf16_kernel<<<(nw1 + 255) / 256, 256, 0, stream>>>(Wqkv,  wqkvb,  nw1);
  cvt_bf16_kernel<<<(nw2 + 255) / 256, 256, 0, stream>>>(Wproj, wprojb, nw2);

  // qkv = x @ W_qkv + b_qkv  -> Q/K/V bf16 [B,H,T,64]
  gemm_bf16_kernel<0><<<dim3(3 * DMODEL / 128, MROWS / 128), 256, 0, stream>>>(
      xb, wqkvb, bqkv, nullptr, Qb, Kbuf, Vbuf, MROWS, 3 * DMODEL, DMODEL);

  // causal flash attention -> attb bf16 [B,T,C]
  attn_kernel<<<dim3(TSEQ / 64, BATCH * NHEAD), 128, 0, stream>>>(Qb, Kbuf, Vbuf, attb);

  // out = attb @ W_proj + b_proj (fp32)
  gemm_bf16_kernel<1><<<dim3(DMODEL / 128, MROWS / 128), 256, 0, stream>>>(
      attb, wprojb, bproj, out, nullptr, nullptr, nullptr, MROWS, DMODEL, DMODEL);
}
